// MixTemporalGNN_SSL_85590108274694
// MI455X (gfx1250) — matrix operand order repaired and checked by reading, not compiled, not run on hardware
//
#include <hip/hip_runtime.h>

// ---------------------------------------------------------------------------
// MixTemporalGNN forward for MI455X (gfx1250, wave32, WMMA + async-to-LDS).
//
// GEMM design:
//  * All weight (B) matrices are converted once per launch to bf16 and
//    pre-transposed to a uniform K-contiguous [N,K] layout in workspace.
//  * 64x64 block tiles, double-buffered LDS: A tile (f32, 8KB) and B tile
//    (bf16, 4KB) are DMA'd with global_load_async_to_lds_b128 (ASYNCcnt,
//    no VGPR round-trip) for tile i+1 while tile i is consumed.
//    One s_wait_asynccnt + one workgroup barrier per K step.
//  * 4 waves/block, each wave -> 16(M) x 64(N): 4 v_wmma_f32_16x16x32_bf16
//    per 32-wide K step; A converted f32->bf16 at fragment build, B fragments
//    are direct 32B v16bf LDS loads (zero conversions on the B path).
//  * LSTM weights (~84 MB bf16) stay resident in the 192 MB L2 across the
//    15 recurrent steps.
// ---------------------------------------------------------------------------

typedef __bf16 bf16_t;
typedef __bf16 v16bf __attribute__((ext_vector_type(16)));
typedef float  v4f   __attribute__((ext_vector_type(4)));
typedef float  v8f   __attribute__((ext_vector_type(8)));

#define N_NODES 61440
#define N_EDGES 491520
#define EMB_D   64
#define HFEATS  128
#define GOUT    512          // 4*HFEATS, hcat width
#define NGRAPH  960
#define BATCH   64
#define POINT   15
#define HID     1024
#define BN_EPS  1e-5f

__device__ __forceinline__ float sigf(float x) { return 1.0f / (1.0f + expf(-x)); }

// ---- gfx1250 async global->LDS copies (ASYNCcnt) ----
__device__ __forceinline__ unsigned lds_off(const void* p) {
    // generic LDS pointer: low 32 bits are the wave-relative LDS byte address
    return (unsigned)(unsigned long long)p;
}
__device__ __forceinline__ void async_ld_b128(unsigned lds, const void* g) {
    asm volatile("global_load_async_to_lds_b128 %0, %1, off"
                 :: "v"(lds), "v"((unsigned long long)g) : "memory");
}
__device__ __forceinline__ void wait_async0() {
    asm volatile("s_wait_asynccnt 0x0" ::: "memory");
}

// ---------------------------------------------------------------------------
// Block-tiled WMMA GEMM: C[M,N] = A[M,K] * Bw^T-layout + bias0 + bias1 (+C).
// A: f32 [M,K] row-major (lda).  Bw: bf16 [N,K] row-major (ldb == K stride),
// i.e. Bw[n*ldb + k] == B(k,n).  M%64==0, N%64==0, K%32==0 (all GEMMs here).
// blockDim = 128 (4 waves). Block tile 64x64, K step 32, double-buffered LDS.
// ---------------------------------------------------------------------------
__global__ __launch_bounds__(128) void wmma_gemm_kernel(
    const float* __restrict__ A, int lda,
    const bf16_t* __restrict__ Bw, int ldb,
    const float* __restrict__ bias0, const float* __restrict__ bias1,
    float* __restrict__ C, int ldc,
    int M, int Nn, int K, int accumulate)
{
    __shared__ float  Af[2][64][32];   // [m][k]  2 x 8 KB
    __shared__ bf16_t Bf[2][64][32];   // [n][k]  2 x 4 KB (K-contiguous rows)

    const int tilesN = Nn >> 6;
    const int tile   = blockIdx.x;
    const int tm     = (tile / tilesN) << 6;
    const int tn     = (tile % tilesN) << 6;

    const int tid  = threadIdx.x;        // 0..127
    const int w    = tid >> 5;           // wave 0..3 -> M sub-rows [w*16, w*16+16)
    const int lane = tid & 31;
    const int half = lane >> 4;          // 0 | 1
    const int l16  = lane & 15;
    const int mrow = (w << 4) + l16;     // row within 64-row tile this lane feeds

    v8f acc[4] = {};

    // issue async fill of tile `k0` into buffer `buf`
    auto fill_tiles = [&](int k0, int buf) {
        // A tile: 64 rows x 32 k f32 = 512 x 16B chunks, 4 per thread
#pragma unroll
        for (int i = 0; i < 4; ++i) {
            const int chunk = tid + (i << 7);        // 0..511
            const int row   = chunk >> 3;            // 0..63
            const int c4    = (chunk & 7) << 2;      // 0,4,..,28
            async_ld_b128(lds_off(&Af[buf][row][c4]),
                          A + (long)(tm + row) * lda + k0 + c4);
        }
        // B tile: 64 rows x 32 k bf16 = 256 x 16B chunks, 2 per thread
#pragma unroll
        for (int i = 0; i < 2; ++i) {
            const int chunk = tid + (i << 7);        // 0..255
            const int n  = chunk >> 2;               // 0..63
            const int k8 = (chunk & 3) << 3;         // 0,8,16,24
            async_ld_b128(lds_off(&Bf[buf][n][k8]),
                          Bw + (long)(tn + n) * ldb + k0 + k8);
        }
    };

    fill_tiles(0, 0);
    const int nIter = K >> 5;
    for (int it = 0; it < nIter; ++it) {
        const int cur = it & 1;
        wait_async0();          // this wave's outstanding async copies landed
        __syncthreads();        // all waves: tile `cur` full, tile `cur^1` free
        if (it + 1 < nIter) fill_tiles((it + 1) << 5, cur ^ 1);
        // speculative prefetch two K tiles ahead (global_prefetch_b8)
        __builtin_prefetch(A + (long)(tm + (tid >> 1)) * lda + (it << 5) + 64, 0, 1);

        // ---- fragments + 4 WMMAs (overlaps with async fill of cur^1) ----
        // A fragment: lane(l16,half) takes K in [16h,16h+16) (matches B);
        // convert f32->bf16 here.
        v4f a0 = *(const v4f*)&Af[cur][mrow][(half << 4) + 0];
        v4f a1 = *(const v4f*)&Af[cur][mrow][(half << 4) + 4];
        v4f a2 = *(const v4f*)&Af[cur][mrow][(half << 4) + 8];
        v4f a3 = *(const v4f*)&Af[cur][mrow][(half << 4) + 12];
        v16bf a;
#pragma unroll
        for (int j = 0; j < 4; ++j) {
            a[j + 0]  = (bf16_t)a0[j];
            a[j + 4]  = (bf16_t)a1[j];
            a[j + 8]  = (bf16_t)a2[j];
            a[j + 12] = (bf16_t)a3[j];
        }
#pragma unroll
        for (int sub = 0; sub < 4; ++sub) {
            // B fragment: col = sub*16 + l16, K = 16*half + j -> one 32B load
            v16bf b = *(const v16bf*)&Bf[cur][(sub << 4) + l16][half << 4];
            acc[sub] = __builtin_amdgcn_wmma_f32_16x16x32_bf16(
                false, a, false, b, (short)0, acc[sub], false, false);
        }
    }

    // ---- epilogue: C/D layout -> element r: row = r + 8*half, col = l16 ----
#pragma unroll
    for (int sub = 0; sub < 4; ++sub) {
        const int col = tn + (sub << 4) + l16;
        float badd = 0.0f;
        if (bias0) badd += bias0[col];
        if (bias1) badd += bias1[col];
#pragma unroll
        for (int r = 0; r < 8; ++r) {
            const int row = tm + (w << 4) + r + (half << 3);
            float v = acc[sub][r] + badd;
            float* cp = C + (long)row * ldc + col;
            *cp = accumulate ? (*cp + v) : v;
        }
    }
}

// ---------------------------------------------------------------------------
// One-shot weight prep: f32 -> bf16, unified [N,K] K-contiguous layout
// ---------------------------------------------------------------------------
// in: [K,N] f32 row-major -> out: [N,K] bf16 (transpose + convert)
__global__ void convT_bf16_kernel(const float* __restrict__ in, bf16_t* __restrict__ out,
                                  int K, int N, int total) {
    int i = blockIdx.x * blockDim.x + threadIdx.x;
    if (i < total) {
        int n = i / K, k = i % K;
        out[i] = (bf16_t)in[(long)k * N + n];
    }
}
// in: [N,K] f32 row-major -> out: [N,K] bf16 (convert only)
__global__ void conv_bf16_kernel(const float* __restrict__ in, bf16_t* __restrict__ out,
                                 long total) {
    long i = (long)blockIdx.x * blockDim.x + threadIdx.x;
    if (i < total) out[i] = (bf16_t)in[i];
}

// ---------------------------------------------------------------------------
// Elementwise / graph kernels
// ---------------------------------------------------------------------------
__global__ void fill_zero_kernel(float* p, long n) {
    long i = (long)blockIdx.x * blockDim.x + threadIdx.x;
    if (i < n) p[i] = 0.0f;
}

__global__ void embed_kernel(const int* __restrict__ feat, const float* __restrict__ emb,
                             float* __restrict__ h, int total) {
    int i = blockIdx.x * blockDim.x + threadIdx.x;
    if (i < total) {
        int n = i >> 6, k = i & 63;
        h[i] = emb[feat[n] * EMB_D + k];
    }
}

__global__ void deg_kernel(const int* __restrict__ dst, float* __restrict__ deg, int e) {
    int i = blockIdx.x * blockDim.x + threadIdx.x;
    if (i < e) atomicAdd(&deg[dst[i]], 1.0f);
}

__global__ void scatter_kernel(const float* __restrict__ h, int ldh,
                               const int* __restrict__ src, const int* __restrict__ dst,
                               float* __restrict__ agg, int din, long total) {
    long i = (long)blockIdx.x * blockDim.x + threadIdx.x;
    if (i < total) {
        int e = (int)(i / din), k = (int)(i % din);
        atomicAdd(&agg[(long)dst[e] * din + k], h[(long)src[e] * ldh + k]);
    }
}

__global__ void norm_agg_kernel(float* __restrict__ agg, const float* __restrict__ deg,
                                int din, long total) {
    long i = (long)blockIdx.x * blockDim.x + threadIdx.x;
    if (i < total) {
        int n = (int)(i / din);
        agg[i] *= 1.0f / fmaxf(deg[n], 1.0f);
    }
}

// PReLU in place + per-feature sum/sumsq (block-reduced before atomics).
__global__ void prelu_stats_kernel(float* __restrict__ y, int ldy,
                                   const float* __restrict__ alpha,
                                   float* __restrict__ stats, int rows) {
    int f  = threadIdx.x;                 // 0..127
    int r0 = blockIdx.x * rows;
    float a = alpha[f], s = 0.0f, s2 = 0.0f;
    for (int r = 0; r < rows; ++r) {
        long idx = (long)(r0 + r) * ldy + f;
        float v = y[idx];
        v = (v >= 0.0f) ? v : a * v;
        y[idx] = v;
        s += v; s2 += v * v;
    }
    atomicAdd(&stats[f], s);
    atomicAdd(&stats[HFEATS + f], s2);
}

__global__ void bn_apply_kernel(float* __restrict__ y, int ldy,
                                const float* __restrict__ stats,
                                const float* __restrict__ gamma,
                                const float* __restrict__ beta, int total) {
    int i = blockIdx.x * blockDim.x + threadIdx.x;
    if (i < total) {
        int n = i / HFEATS, f = i % HFEATS;
        float mu  = stats[f] * (1.0f / N_NODES);
        float var = stats[HFEATS + f] * (1.0f / N_NODES) - mu * mu;
        long idx = (long)n * ldy + f;
        y[idx] = gamma[f] * (y[idx] - mu) * rsqrtf(var + BN_EPS) + beta[f];
    }
}

__global__ void graph_mean_kernel(const float* __restrict__ hcat, float* __restrict__ gm,
                                  int total) {
    int i = blockIdx.x * blockDim.x + threadIdx.x;
    if (i < total) {
        int g = i / GOUT, f = i % GOUT;
        const float* p = hcat + (long)g * 64 * GOUT + f;
        float s = 0.0f;
        for (int k = 0; k < 64; ++k) s += p[(long)k * GOUT];
        gm[i] = s * (1.0f / 64.0f);
    }
}

__global__ void prelu_point_kernel(float* __restrict__ z, const float* __restrict__ a,
                                   int total) {
    int i = blockIdx.x * blockDim.x + threadIdx.x;
    if (i < total) {
        int p = (i / GOUT) % POINT;
        float v = z[i];
        z[i] = (v >= 0.0f) ? v : a[p] * v;
    }
}

// sigmoid gates -> write straight into LSTM [t][b][1024] layout (transpose fused)
__global__ void gate_kernel(const float* __restrict__ zf1, const float* __restrict__ zf2,
                            const float* __restrict__ hx, const float* __restrict__ hy,
                            float* __restrict__ xbuf, int total) {
    int i = blockIdx.x * blockDim.x + threadIdx.x;
    if (i < total) {
        int r = i / GOUT, f = i % GOUT;
        int b = r / POINT, t = r % POINT;
        long base = ((long)t * BATCH + b) * (2 * GOUT);
        xbuf[base + f]        = sigf(zf1[i]) * hy[i];
        xbuf[base + GOUT + f] = sigf(zf2[i]) * hx[i];
    }
}

__global__ void lstm_cell_kernel(const float* __restrict__ g, float* __restrict__ h,
                                 float* __restrict__ c, float* __restrict__ yout,
                                 int dirOff) {
    int i = blockIdx.x * blockDim.x + threadIdx.x;   // 64*1024
    if (i < BATCH * HID) {
        int b = i >> 10, j = i & (HID - 1);
        const float* gb = g + (long)b * 4 * HID;
        float gi = gb[j], gf = gb[HID + j], gg = gb[2 * HID + j], go = gb[3 * HID + j];
        float cn = sigf(gf) * c[i] + sigf(gi) * tanhf(gg);
        float hn = sigf(go) * tanhf(cn);
        c[i] = cn; h[i] = hn;
        yout[(long)b * (2 * HID) + dirOff + j] = hn;
    }
}

__global__ void rnncat_kernel(const float* __restrict__ hb, const float* __restrict__ hf,
                              float* __restrict__ out, int total) {
    int i = blockIdx.x * blockDim.x + threadIdx.x;   // 64*2048
    if (i < total) {
        int b = i / (2 * HID), j = i % (2 * HID);
        out[i] = (j < HID) ? hb[b * HID + j] : hf[b * HID + (j - HID)];
    }
}

__global__ void prelu_vec_kernel(const float* __restrict__ x, const float* __restrict__ a,
                                 float* __restrict__ out, int total) {
    int i = blockIdx.x * blockDim.x + threadIdx.x;
    if (i < total) {
        float v = x[i];
        int f = i % GOUT;
        out[i] = (v >= 0.0f) ? v : a[f] * v;
    }
}

// ---------------------------------------------------------------------------
// Host orchestration
// ---------------------------------------------------------------------------
static inline unsigned blocks_for(long n, int bs) { return (unsigned)((n + bs - 1) / bs); }

extern "C" void kernel_launch(void* const* d_in, const int* in_sizes, int n_in,
                              void* d_out, int out_size, void* d_ws, size_t ws_size,
                              hipStream_t stream) {
    (void)in_sizes; (void)n_in; (void)out_size;

    const int* header_feat = (const int*)d_in[0];
    const int* header_src  = (const int*)d_in[1];
    const int* header_dst  = (const int*)d_in[2];
    const int* payload_feat= (const int*)d_in[3];
    const int* payload_src = (const int*)d_in[4];
    const int* payload_dst = (const int*)d_in[5];
    // d_in[6] = graph_id (arange/64; mean pool uses the fixed 64-node layout)

    void* const* HP = d_in + 7;    // gcn header params: emb + 4 x {wS,wN,b,prelu,gamma,beta}
    void* const* PP = d_in + 32;   // gcn payload params
    void* const* FP = d_in + 57;   // f1{w1,b1,a,w2,b2}, f2{...}
    void* const* LP = d_in + 67;   // l0{fwd(wih,whh,bih,bhh),bwd},l1{...}
    void* const* EP = d_in + 83;   // encoder {w,b,prelu}

    // ---- workspace layout (float-granular allocator, 256B-aligned blocks) ----
    float* ws = (float*)d_ws;
    size_t off = 0;
    auto alloc  = [&](size_t n) { float* p = ws + off; off += (n + 63) & ~(size_t)63; return p; };
    auto allocb = [&](size_t n) { bf16_t* p = (bf16_t*)(ws + off);
                                  off += ((n + 1) / 2 + 63) & ~(size_t)63; return p; };
    float* hcatH  = alloc((size_t)N_NODES * GOUT);
    float* hcatP  = alloc((size_t)N_NODES * GOUT);
    float* hemb   = alloc((size_t)N_NODES * EMB_D);
    float* agg    = alloc((size_t)N_NODES * HFEATS);
    float* deg    = alloc(N_NODES);
    float* stats  = alloc(2 * HFEATS);
    float* hx     = alloc((size_t)NGRAPH * GOUT);
    float* hy     = alloc((size_t)NGRAPH * GOUT);
    float* ztmp   = alloc((size_t)NGRAPH * GOUT);
    float* zf1    = alloc((size_t)NGRAPH * GOUT);
    float* zf2    = alloc((size_t)NGRAPH * GOUT);
    float* xbuf0  = alloc((size_t)POINT * BATCH * 2 * GOUT);     // [15,64,1024]
    float* ybuf0  = alloc((size_t)POINT * BATCH * 2 * HID);      // [15,64,2048]
    float* ybuf1  = alloc((size_t)POINT * BATCH * 2 * HID);
    float* hF     = alloc(BATCH * HID);
    float* cF     = alloc(BATCH * HID);
    float* hB     = alloc(BATCH * HID);
    float* cB     = alloc(BATCH * HID);
    float* gbuf   = alloc((size_t)BATCH * 4 * HID);
    float* rncat  = alloc((size_t)BATCH * 2 * HID);
    float* enctmp = alloc((size_t)BATCH * GOUT);
    // bf16 weights, unified [N,K] layout
    bf16_t* gwT[2][4][2];            // [gcn][layer][self|neigh], K=din, N=128
    for (int g = 0; g < 2; ++g)
        for (int l = 0; l < 4; ++l) {
            int din = l ? HFEATS : EMB_D;
            gwT[g][l][0] = allocb((size_t)HFEATS * din);
            gwT[g][l][1] = allocb((size_t)HFEATS * din);
        }
    bf16_t* fwT[2][2];               // [filter][w1|w2], 512x512
    for (int f = 0; f < 2; ++f)
        for (int m = 0; m < 2; ++m) fwT[f][m] = allocb((size_t)GOUT * GOUT);
    bf16_t* lwT[2][2][2];            // [layer][dir][wih|whh]
    for (int l = 0; l < 2; ++l)
        for (int d = 0; d < 2; ++d) {
            int xdim = l ? 2 * HID : 2 * GOUT;
            lwT[l][d][0] = allocb((size_t)4 * HID * xdim);
            lwT[l][d][1] = allocb((size_t)4 * HID * HID);
        }
    bf16_t* encwT = allocb((size_t)GOUT * 2 * HID);   // [512, 2048]
    if (off * sizeof(float) > ws_size) return;  // workspace too small: bail deterministically

    auto fill0 = [&](float* p, long n) {
        fill_zero_kernel<<<blocks_for(n, 256), 256, 0, stream>>>(p, n);
    };
    auto convT = [&](const float* in, bf16_t* out, int K, int N) {   // in [K,N] -> out [N,K]
        convT_bf16_kernel<<<blocks_for((long)K * N, 256), 256, 0, stream>>>(in, out, K, N, K * N);
    };
    auto convS = [&](const float* in, bf16_t* out, long total) {     // in [N,K] -> out [N,K]
        conv_bf16_kernel<<<blocks_for(total, 256), 256, 0, stream>>>(in, out, total);
    };
    auto gemm = [&](const float* A, int lda, const bf16_t* Bw,
                    const float* b0, const float* b1, float* C, int ldc,
                    int M, int Nn, int K, int acc) {
        wmma_gemm_kernel<<<dim3((unsigned)((M / 64) * (Nn / 64))), dim3(128), 0, stream>>>(
            A, lda, Bw, K, b0, b1, C, ldc, M, Nn, K, acc);
    };

    // ---- weight prep (once per launch) ----
    for (int g = 0; g < 2; ++g) {
        void* const* prm = g ? PP : HP;
        for (int l = 0; l < 4; ++l) {
            int din = l ? HFEATS : EMB_D;
            convT((const float*)prm[1 + l * 6 + 0], gwT[g][l][0], din, HFEATS);
            convT((const float*)prm[1 + l * 6 + 1], gwT[g][l][1], din, HFEATS);
        }
    }
    for (int f = 0; f < 2; ++f) {
        convT((const float*)FP[f * 5 + 0], fwT[f][0], GOUT, GOUT);
        convT((const float*)FP[f * 5 + 3], fwT[f][1], GOUT, GOUT);
    }
    for (int l = 0; l < 2; ++l)
        for (int d = 0; d < 2; ++d) {
            int xdim = l ? 2 * HID : 2 * GOUT;
            convS((const float*)LP[l * 8 + d * 4 + 0], lwT[l][d][0], (long)4 * HID * xdim);
            convS((const float*)LP[l * 8 + d * 4 + 1], lwT[l][d][1], (long)4 * HID * HID);
        }
    convT((const float*)EP[0], encwT, 2 * HID, GOUT);

    // ---- GCN (x2) ----
    auto run_gcn = [&](const int* feat, const int* src, const int* dst,
                       void* const* prm, bf16_t* (*wt)[2], float* hcat) {
        embed_kernel<<<blocks_for((long)N_NODES * EMB_D, 256), 256, 0, stream>>>(
            feat, (const float*)prm[0], hemb, N_NODES * EMB_D);
        fill0(deg, N_NODES);
        deg_kernel<<<blocks_for(N_EDGES, 256), 256, 0, stream>>>(dst, deg, N_EDGES);

        const float* hin = hemb; int lda = EMB_D; int din = EMB_D;
        for (int l = 0; l < 4; ++l) {
            const float* bias   = (const float*)prm[1 + l * 6 + 2];
            const float* alpha  = (const float*)prm[1 + l * 6 + 3];
            const float* gamma  = (const float*)prm[1 + l * 6 + 4];
            const float* beta   = (const float*)prm[1 + l * 6 + 5];
            float* hout = hcat + l * HFEATS;   // slice of [N,512]

            fill0(agg, (long)N_NODES * din);
            long tot = (long)N_EDGES * din;
            scatter_kernel<<<blocks_for(tot, 256), 256, 0, stream>>>(
                hin, lda, src, dst, agg, din, tot);
            norm_agg_kernel<<<blocks_for((long)N_NODES * din, 256), 256, 0, stream>>>(
                agg, deg, din, (long)N_NODES * din);

            gemm(hin, lda, wt[l][0], bias, nullptr, hout, GOUT, N_NODES, HFEATS, din, 0);
            gemm(agg, din, wt[l][1], nullptr, nullptr, hout, GOUT, N_NODES, HFEATS, din, 1);

            fill0(stats, 2 * HFEATS);
            prelu_stats_kernel<<<N_NODES / 32, HFEATS, 0, stream>>>(hout, GOUT, alpha, stats, 32);
            bn_apply_kernel<<<blocks_for((long)N_NODES * HFEATS, 256), 256, 0, stream>>>(
                hout, GOUT, stats, gamma, beta, N_NODES * HFEATS);

            hin = hout; lda = GOUT; din = HFEATS;
        }
    };
    run_gcn(header_feat,  header_src,  header_dst,  HP, gwT[0], hcatH);
    run_gcn(payload_feat, payload_src, payload_dst, PP, gwT[1], hcatP);

    graph_mean_kernel<<<blocks_for((long)NGRAPH * GOUT, 256), 256, 0, stream>>>(
        hcatH, hx, NGRAPH * GOUT);
    graph_mean_kernel<<<blocks_for((long)NGRAPH * GOUT, 256), 256, 0, stream>>>(
        hcatP, hy, NGRAPH * GOUT);

    // ---- filters: zf = Lin2(PReLU_point(Lin1(x))) ----
    auto run_filter = [&](const float* x, void* const* fp, bf16_t* const* wt, float* zout) {
        const float* b1 = (const float*)fp[1];
        const float* a  = (const float*)fp[2];
        const float* b2 = (const float*)fp[4];
        gemm(x, GOUT, wt[0], b1, nullptr, ztmp, GOUT, NGRAPH, GOUT, GOUT, 0);
        prelu_point_kernel<<<blocks_for((long)NGRAPH * GOUT, 256), 256, 0, stream>>>(
            ztmp, a, NGRAPH * GOUT);
        gemm(ztmp, GOUT, wt[1], b2, nullptr, zout, GOUT, NGRAPH, GOUT, GOUT, 0);
    };
    run_filter(hx, FP + 0, fwT[0], zf1);
    run_filter(hy, FP + 5, fwT[1], zf2);
    gate_kernel<<<blocks_for((long)NGRAPH * GOUT, 256), 256, 0, stream>>>(
        zf1, zf2, hx, hy, xbuf0, NGRAPH * GOUT);

    // ---- BiLSTM (2 layers x 2 dirs x 15 steps) ----
    const float* xin = xbuf0; int xdim = 2 * GOUT;   // 1024
    float* youts[2] = { ybuf0, ybuf1 };
    for (int l = 0; l < 2; ++l) {
        for (int d = 0; d < 2; ++d) {
            const float* bih = (const float*)LP[l * 8 + d * 4 + 2];
            const float* bhh = (const float*)LP[l * 8 + d * 4 + 3];
            float* h = d ? hB : hF; float* c = d ? cB : cF;
            fill0(h, BATCH * HID);
            fill0(c, BATCH * HID);
            for (int step = 0; step < POINT; ++step) {
                int t = d ? (POINT - 1 - step) : step;
                const float* xt = xin + (long)t * BATCH * xdim;
                gemm(xt, xdim, lwT[l][d][0], bih, bhh, gbuf, 4 * HID,
                     BATCH, 4 * HID, xdim, 0);                       // g = x W_ih^T + b
                gemm(h, HID, lwT[l][d][1], nullptr, nullptr, gbuf, 4 * HID,
                     BATCH, 4 * HID, HID, 1);                        // g += h W_hh^T
                float* yt = youts[l] + (long)t * BATCH * 2 * HID;
                lstm_cell_kernel<<<blocks_for(BATCH * HID, 256), 256, 0, stream>>>(
                    gbuf, h, c, yt, d ? HID : 0);
            }
        }
        xin = youts[l]; xdim = 2 * HID;
    }

    // ---- encoder: PReLU(cat(hb, hf) @ W + b) ----
    rnncat_kernel<<<blocks_for(BATCH * 2 * HID, 256), 256, 0, stream>>>(
        hB, hF, rncat, BATCH * 2 * HID);
    gemm(rncat, 2 * HID, encwT, (const float*)EP[1], nullptr,
         enctmp, GOUT, BATCH, GOUT, 2 * HID, 0);
    prelu_vec_kernel<<<blocks_for(BATCH * GOUT, 256), 256, 0, stream>>>(
        enctmp, (const float*)EP[2], (float*)d_out, BATCH * GOUT);
}
